// scattering_2d_partial_cov_21174188769981
// MI455X (gfx1250) — matrix operand order repaired and checked by reading, not compiled, not run on hardware
//
#include <hip/hip_runtime.h>
#include <hip/hip_fp16.h>

// ---- problem constants ----
#define NSIDE  512
#define NPIX   262144        // 512*512
#define NFILT  257           // 1 + K*J
#define MROWS  288           // padded to 9*32 for 32x32 wave tiles
#define GDIM   288
#define FCHUNK 16            // filters per FFT chunk (intermediate = 33.5 MB)
#define KSTEPS 128           // K-steps (of 32) per gram wave; gridDim.z = 8192/KSTEPS = 64

typedef __attribute__((ext_vector_type(16))) _Float16 v16h;
typedef __attribute__((ext_vector_type(8)))  _Float16 v8h;
typedef __attribute__((ext_vector_type(8)))  float    v8f;

// ---------------- 512-point in-place DIT IFFT (e^{+i}) in LDS ----------------
__device__ __forceinline__ void fft512_inv(float2* lds, int t) {
#pragma unroll
  for (int s = 0; s < 9; ++s) {
    __syncthreads();                    // first iter also covers the scatter fill
    const int half = 1 << s;
    const int pos  = t & (half - 1);
    const int i0   = ((t >> s) << (s + 1)) + pos;
    const int i1   = i0 + half;
    const float ang = 6.2831853071795864769f * (float)pos / (float)(half << 1);
    float sw, cw;
    __sincosf(ang, &sw, &cw);           // w = e^{+i ang} (inverse transform)
    const float2 v0 = lds[i0];
    const float2 v1 = lds[i1];
    const float tr = cw * v1.x - sw * v1.y;
    const float ti = cw * v1.y + sw * v1.x;
    lds[i0] = make_float2(v0.x + tr, v0.y + ti);
    lds[i1] = make_float2(v0.x - tr, v0.y - ti);
  }
}

// Pass 1: y_hat = x_hat * psi_hat (complex), row IFFT, write W[chunk_f][y][x]
__global__ __launch_bounds__(256) void pass1_rows(
    const float* __restrict__ x_hat,   // [512,512,2]
    const float* __restrict__ psir,    // [257,512,512]
    const float* __restrict__ psii,
    float2* __restrict__ W,            // [FCHUNK,512,512]
    int fbase) {
  __shared__ float2 lds[NSIDE];
  const int yrow = blockIdx.x;
  const int fl   = blockIdx.y;               // chunk-local filter
  const int f    = fbase + fl;
  const int t    = threadIdx.x;              // 0..255
  const size_t plane = (size_t)f * NPIX + (size_t)yrow * NSIDE;
  const size_t xrow  = (size_t)yrow * NSIDE;
#pragma unroll
  for (int e = 0; e < 2; ++e) {
    const int x = t + e * 256;
    const float xr = x_hat[(xrow + x) * 2 + 0];
    const float xi = x_hat[(xrow + x) * 2 + 1];
    const float pr = psir[plane + x];
    const float pi = psii[plane + x];
    const int rb = __brev((unsigned)x) >> 23;   // 9-bit bit-reversal
    lds[rb] = make_float2(xr * pr - xi * pi, xr * pi + xi * pr);
  }
  fft512_inv(lds, t);
  __syncthreads();
  float2* Wrow = W + (size_t)fl * NPIX + (size_t)yrow * NSIDE;
  Wrow[t]       = lds[t];
  Wrow[t + 256] = lds[t + 256];
}

// Pass 2: column IFFT, scale 1/N^2, modulus -> f16 row of M (pixel order = col-major, consistent)
__global__ __launch_bounds__(256) void pass2_cols(
    const float2* __restrict__ W,      // [FCHUNK,512,512]
    _Float16* __restrict__ Mb,         // [MROWS, NPIX]
    int fbase) {
  __shared__ float2 lds[NSIDE];
  const int xcol = blockIdx.x;
  const int fl   = blockIdx.y;
  const int f    = fbase + fl;
  const int t    = threadIdx.x;
  const float2* Wp = W + (size_t)fl * NPIX;
#pragma unroll
  for (int e = 0; e < 2; ++e) {
    const int y = t + e * 256;
    const int rb = __brev((unsigned)y) >> 23;
    lds[rb] = Wp[(size_t)y * NSIDE + xcol];
  }
  fft512_inv(lds, t);
  __syncthreads();
  const float scale = 1.0f / 262144.0f;       // ifft2 normalization
  _Float16* Mrow = Mb + (size_t)f * NPIX + (size_t)xcol * NSIDE;
#pragma unroll
  for (int e = 0; e < 2; ++e) {
    const int y = t + e * 256;
    const float2 v = lds[y];
    const float re = v.x * scale, im = v.y * scale;
    Mrow[y] = (_Float16)sqrtf(re * re + im * im + 1e-8f);
  }
}

// Zero G accumulator and the zero-pad rows 257..287 of M (deterministic each call)
__global__ void zero_init(float* __restrict__ G, _Float16* __restrict__ Mb) {
  const size_t id = (size_t)blockIdx.x * blockDim.x + threadIdx.x;
  const size_t stride = (size_t)gridDim.x * blockDim.x;
  for (size_t i = id; i < (size_t)GDIM * GDIM; i += stride) G[i] = 0.0f;
  _Float16* pad = Mb + (size_t)NFILT * NPIX;
  const size_t padN = (size_t)(MROWS - NFILT) * NPIX;
  for (size_t i = id; i < padN; i += stride) pad[i] = (_Float16)0.0f;
}

#define WMMA_F16(A, B, C) \
  __builtin_amdgcn_wmma_f32_16x16x32_f16(false, (A), false, (B), (short)0, (C), false, false)

// G += M * M^T over a K chunk. One wave (32 threads) per 32x32 output tile.
// grid = (9, 9, 64); block = 32.
__global__ __launch_bounds__(32) void gram_wmma(
    const _Float16* __restrict__ Mb,   // [MROWS, NPIX] f16
    float* __restrict__ G) {           // [288, 288] f32, atomically accumulated
  const int lane = threadIdx.x;
  const int g = lane >> 4;             // K half-group
  const int r = lane & 15;             // A: row-in-tile, B: col-in-tile
  const _Float16* a0 = Mb + (size_t)(blockIdx.x * 32 + r) * NPIX;
  const _Float16* a1 = a0 + (size_t)16 * NPIX;
  const _Float16* b0 = Mb + (size_t)(blockIdx.y * 32 + r) * NPIX + (size_t)g * 16;
  const _Float16* b1 = b0 + (size_t)16 * NPIX;
  const int aoff = g * 16;             // A fragment K-slot mapping: k = 16*g + s (matches B contiguous map)
  size_t k = (size_t)blockIdx.z * ((size_t)KSTEPS * 32);

  v8f c00 = {}, c01 = {}, c10 = {}, c11 = {};
#pragma unroll 2
  for (int s = 0; s < KSTEPS; ++s, k += 32) {
    const v8h a0lo = *(const v8h*)(a0 + k + aoff);
    const v8h a0hi = *(const v8h*)(a0 + k + 8 + aoff);
    const v8h a1lo = *(const v8h*)(a1 + k + aoff);
    const v8h a1hi = *(const v8h*)(a1 + k + 8 + aoff);
    const v16h A0 = __builtin_shufflevector(a0lo, a0hi, 0,1,2,3,4,5,6,7,8,9,10,11,12,13,14,15);
    const v16h A1 = __builtin_shufflevector(a1lo, a1hi, 0,1,2,3,4,5,6,7,8,9,10,11,12,13,14,15);
    const v16h B0 = *(const v16h*)(b0 + k);
    const v16h B1 = *(const v16h*)(b1 + k);
    c00 = WMMA_F16(A0, B0, c00);
    c01 = WMMA_F16(A0, B1, c01);
    c10 = WMMA_F16(A1, B0, c10);
    c11 = WMMA_F16(A1, B1, c11);
  }
  // C/D layout: VGPR v, lanes 0-15 -> M=v, lanes 16-31 -> M=v+8; N = lane%16
  const int rr0 = blockIdx.x * 32 + g * 8;
  const int cc  = blockIdx.y * 32 + r;
#pragma unroll
  for (int v = 0; v < 8; ++v) {
    const int rr = rr0 + v;
    atomicAdd(&G[(size_t)rr * GDIM + cc],               c00[v]);
    atomicAdd(&G[(size_t)rr * GDIM + cc + 16],          c01[v]);
    atomicAdd(&G[(size_t)(rr + 16) * GDIM + cc],        c10[v]);
    atomicAdd(&G[(size_t)(rr + 16) * GDIM + cc + 16],   c11[v]);
  }
}

// Assemble the 4353-element output in the reference's exact ordering.
// block b = i*16+j handles that (i,j)'s [E2, Ea, C1[i,i+1:,j], C2[i,j,j+1:]] slice.
__global__ __launch_bounds__(32) void assemble(const float* __restrict__ G,
                                               float* __restrict__ out) {
  const int b = blockIdx.x;            // 0..255
  const int i = b >> 4, j = b & 15;
  const int t = threadIdx.x;           // 0..31
  const float inv_nn = 1.0f / 262144.0f;
  const int base = 1 + 392 * i - 8 * i * (i - 1) + j * (32 - i) - (j * (j - 1)) / 2;
  const int len  = 32 - i - j;
  const int f    = 1 + i * 16 + j;
  if (b == 0 && t == 0) out[0] = G[0] * inv_nn;            // s0
  if (t < len) {
    float val;
    if (t == 0)      val = G[(size_t)f * GDIM + f];        // E2
    else if (t == 1) val = G[(size_t)f * GDIM + 0];        // Ea
    else if (t < 2 + (15 - i)) {                           // C1: l = i+1..15
      const int l = i + 1 + (t - 2);
      val = G[(size_t)f * GDIM + (1 + l * 16 + j)];
    } else {                                               // C2: l = j+1..15
      const int l = j + 1 + (t - (2 + 15 - i));
      val = G[(size_t)f * GDIM + (1 + i * 16 + l)];
    }
    out[base + t] = val * inv_nn;
  }
}

extern "C" void kernel_launch(void* const* d_in, const int* in_sizes, int n_in,
                              void* d_out, int out_size, void* d_ws, size_t ws_size,
                              hipStream_t stream) {
  const float* x_hat = (const float*)d_in[0];
  const float* psir  = (const float*)d_in[1];
  const float* psii  = (const float*)d_in[2];
  float* out = (float*)d_out;

  // workspace layout (all 256B aligned):
  //   G  : 288*288*4            =    331,776 B
  //   Mb : 288*262144*2         = 150,994,944 B
  //   W  : FCHUNK*262144*8      =  33,554,432 B   -> total ~184.9 MB
  char* ws = (char*)d_ws;
  float*    G  = (float*)ws;
  _Float16* Mb = (_Float16*)(ws + 331776);
  float2*   W  = (float2*)(ws + 331776 + 150994944ull);

  zero_init<<<4096, 256, 0, stream>>>(G, Mb);

  for (int fb = 0; fb < NFILT; fb += FCHUNK) {
    const int nf = (NFILT - fb < FCHUNK) ? (NFILT - fb) : FCHUNK;
    pass1_rows<<<dim3(NSIDE, nf), 256, 0, stream>>>(x_hat, psir, psii, W, fb);
    pass2_cols<<<dim3(NSIDE, nf), 256, 0, stream>>>(W, Mb, fb);
  }

  gram_wmma<<<dim3(9, 9, 64), 32, 0, stream>>>(Mb, G);
  assemble<<<256, 32, 0, stream>>>(G, out);
}
